// SparseLocalAttention_50319836840199
// MI455X (gfx1250) — matrix operand order repaired and checked by reading, not compiled, not run on hardware
//
#include <hip/hip_runtime.h>

typedef __bf16 bf16_t;
typedef __attribute__((ext_vector_type(16))) bf16_t v16bf;
typedef __attribute__((ext_vector_type(8)))  bf16_t v8bf;
typedef __attribute__((ext_vector_type(2)))  bf16_t v2bf;
typedef __attribute__((ext_vector_type(8)))  float  v8f;
typedef __attribute__((ext_vector_type(4)))  float  v4f;

#define S_LEN 4096
#define D_DIM 512
#define BATCH 4
#define WIN   10
#define NEGC  1000000000.0f
#define TJ    64            // keys per mainloop tile
#define NWAVE 4
#define BLOCK (NWAVE * 32)
#define MT    16            // query rows per block
#define NTILE (S_LEN / TJ)  // 64 mainloop iterations

// ---- LDS layout (bf16 element offsets), double-buffered K/V tiles ----
#define Q_ELEMS  (MT * D_DIM)          // 8192
#define K_ELEMS  (TJ * D_DIM)          // 32768 (64KB)
#define V_ELEMS  (D_DIM * TJ)          // 32768 (64KB)
#define P_ELEMS  (MT * TJ)             // 1024
#define Q_OFF   0
#define K0_OFF  (Q_OFF + Q_ELEMS)
#define V0_OFF  (K0_OFF + 2 * K_ELEMS)
#define P_OFF   (V0_OFF + 2 * V_ELEMS)
#define RED_BYTE_OFF ((P_OFF + P_ELEMS) * 2)
#define SMEM_BYTES (RED_BYTE_OFF + 2 * NWAVE * MT * 4)   // ~275KB (<= 320KB/WG on CDNA5)

// one-time fp32 -> bf16 conversion of the whole input into d_ws
__global__ __launch_bounds__(256)
void cvt_to_bf16(const float* __restrict__ in, bf16_t* __restrict__ out16) {
  int i = (blockIdx.x * 256 + threadIdx.x) * 8;
  v4f a = *(const v4f*)(in + i);
  v4f b = *(const v4f*)(in + i + 4);
  v8bf o;
  o[0] = (bf16_t)a[0]; o[1] = (bf16_t)a[1]; o[2] = (bf16_t)a[2]; o[3] = (bf16_t)a[3];
  o[4] = (bf16_t)b[0]; o[5] = (bf16_t)b[1]; o[6] = (bf16_t)b[2]; o[7] = (bf16_t)b[3];
  *(v8bf*)(out16 + i) = o;
}

// async linear copy: n16 * 16 bytes per thread, contiguous global -> LDS
// (GLOBAL_LOAD_ASYNC_TO_LDS_B128, tracked by ASYNCcnt)
__device__ __forceinline__ void async_copy_lin(uint32_t lds_base,
                                               const bf16_t* gbase,
                                               int tid, int n16) {
#pragma unroll
  for (int c = 0; c < 32; ++c) {
    if (c < n16) {
      uint32_t off = (uint32_t)(c * BLOCK + tid) * 16u;
      asm volatile("global_load_async_to_lds_b128 %0, %1, %2"
                   :: "v"(lds_base + off), "v"(off), "s"(gbase)
                   : "memory");
    }
  }
}

__device__ __forceinline__ void wait_async_zero() {
  asm volatile("s_wait_asynccnt 0x0" ::: "memory");
}

// build the [d][key] transposed V tile from the bf16 global tile
__device__ __forceinline__ void fill_vt(bf16_t* vt, const bf16_t* g, int tid) {
  for (int it = 0; it < (TJ / 2) * (D_DIM / 8) / BLOCK; ++it) {
    int idx = it * BLOCK + tid;
    int r2  = idx >> 6;           // key pair 0..31
    int c8  = idx & 63;           // d chunk 0..63
    v8bf a = *(const v8bf*)(g + (size_t)(2 * r2)     * D_DIM + c8 * 8);
    v8bf c = *(const v8bf*)(g + (size_t)(2 * r2 + 1) * D_DIM + c8 * 8);
#pragma unroll
    for (int t = 0; t < 8; ++t) {
      v2bf p; p[0] = a[t]; p[1] = c[t];
      *(v2bf*)(vt + (c8 * 8 + t) * TJ + 2 * r2) = p;
    }
  }
}

__global__ __launch_bounds__(BLOCK)
void sla_flash_wmma(const bf16_t* __restrict__ in16, float* __restrict__ out) {
  extern __shared__ __align__(256) char smem[];
  bf16_t* smb = (bf16_t*)smem;
  bf16_t* Qs = smb + Q_OFF;                    // [MT][D]  (A of QK^T)
  bf16_t* Ps = smb + P_OFF;                    // [MT][TJ] (A of PV)
  float* RedM = (float*)(smem + RED_BYTE_OFF); // [NWAVE][MT]
  float* RedS = RedM + NWAVE * MT;             // [NWAVE][MT]
  // double-buffered tiles, selected arithmetically (no LDS-pointer arrays!)
  //   Ks(buf) = smb + K0_OFF + buf * K_ELEMS   [TJ][D]  (B of QK^T)
  //   Vt(buf) = smb + V0_OFF + buf * V_ELEMS   [D][TJ]  (B of PV)

  const int tid  = threadIdx.x;
  const int wave = tid >> 5;
  const int lane = tid & 31;
  const int hi   = lane >> 4;   // lane-half: C rows v + 8*hi
  const int ln   = lane & 15;   // lane-in-half: C column / A,B row

  const int b  = blockIdx.x >> 8;       // 256 row tiles per batch
  const int mt = blockIdx.x & 255;
  const int i0 = mt * MT;

  const bf16_t* inb  = in16 + (size_t)b * S_LEN * D_DIM;
  float*        outb = out  + (size_t)b * S_LEN * D_DIM;

  // ---- prologue: async-stage Q tile and K tile 0; build Vt tile 0 ----
  async_copy_lin((uint32_t)(uintptr_t)Qs, inb + (size_t)i0 * D_DIM, tid,
                 Q_ELEMS * 2 / (BLOCK * 16));                       // 8 x b128
  async_copy_lin((uint32_t)(uintptr_t)(smb + K0_OFF), inb, tid,
                 K_ELEMS * 2 / (BLOCK * 16));                       // 32 x b128
  fill_vt(smb + V0_OFF, inb, tid);

  // per-lane flash state for the 8 C-layout rows (row = v + 8*hi)
  float m_run[8], l_run[8];
  v8f o[8];   // this wave's 16x128 fp32 output slice (d cols wave*128..+127)
#pragma unroll
  for (int v = 0; v < 8; ++v) { m_run[v] = -3.0e38f; l_run[v] = 0.0f; }
#pragma unroll
  for (int t = 0; t < 8; ++t)
#pragma unroll
    for (int v = 0; v < 8; ++v) o[t][v] = 0.0f;

  for (int i = 0; i < NTILE; ++i) {
    const int cur = i & 1, nxt = cur ^ 1;
    const int j0 = i * TJ;
    bf16_t* KsCur = smb + K0_OFF + cur * K_ELEMS;
    bf16_t* VtCur = smb + V0_OFF + cur * V_ELEMS;

    wait_async_zero();     // this wave's async copies for tile i have landed
    __syncthreads();       // tile i visible to all; buf[nxt] reads (iter i-1) done

    // ---- kick off tile i+1 staging; overlaps with the WMMA work below ----
    if (i + 1 < NTILE) {
      const bf16_t* gn = inb + (size_t)(j0 + TJ) * D_DIM;
      async_copy_lin((uint32_t)(uintptr_t)(smb + K0_OFF + nxt * K_ELEMS), gn,
                     tid, K_ELEMS * 2 / (BLOCK * 16));
      fill_vt(smb + V0_OFF + nxt * V_ELEMS, gn, tid);
    }

    // ---- QK^T: 16x16 score chunk for keys [j0 + wave*16, +16) over D=512 ----
    v8f c;
#pragma unroll
    for (int v = 0; v < 8; ++v) c[v] = 0.0f;
    const int keyBase = wave * 16;
    const bf16_t* qrow = Qs + ln * D_DIM;                 // A: M = ln
    const bf16_t* krow = KsCur + (keyBase + ln) * D_DIM;  // B: N = ln
#pragma unroll
    for (int ks = 0; ks < D_DIM / 32; ++ks) {
      v8bf a0 = *(const v8bf*)(qrow + ks * 32 + hi * 16);
      v8bf a1 = *(const v8bf*)(qrow + ks * 32 + hi * 16 + 8);
      v16bf A = __builtin_shufflevector(a0, a1,
          0,1,2,3,4,5,6,7,8,9,10,11,12,13,14,15);
      v16bf Bv = *(const v16bf*)(krow + ks * 32 + hi * 16);
      c = __builtin_amdgcn_wmma_f32_16x16x32_bf16(false, A, false, Bv,
                                                  (short)0, c, false, false);
    }

    // ---- logits: scores * (mask - 1e9), faithful to reference ----
    const int jc = j0 + keyBase + ln;
    float lm[8];
#pragma unroll
    for (int v = 0; v < 8; ++v) {
      int irow = i0 + v + hi * 8;
      bool band = (unsigned)(jc - irow + WIN) <= (unsigned)(2 * WIN);
      float fac = (band ? 1.0f : 0.0f) - NEGC;
      lm[v] = c[v] * fac;
    }

    // ---- wave-local row max over the 16 lanes of each half ----
    float rmx[8];
#pragma unroll
    for (int v = 0; v < 8; ++v) rmx[v] = lm[v];
#pragma unroll
    for (int m = 1; m < 16; m <<= 1)
#pragma unroll
      for (int v = 0; v < 8; ++v)
        rmx[v] = fmaxf(rmx[v], __shfl_xor(rmx[v], m, 32));
    if (ln == 0) {
#pragma unroll
      for (int v = 0; v < 8; ++v) RedM[wave * MT + v + hi * 8] = rmx[v];
    }
    __syncthreads();

    // ---- cross-wave max, online rescale factors ----
    float alpha[8];
#pragma unroll
    for (int v = 0; v < 8; ++v) {
      int row = v + hi * 8;
      float tm = RedM[row];
#pragma unroll
      for (int w = 1; w < NWAVE; ++w) tm = fmaxf(tm, RedM[w * MT + row]);
      float mnew = fmaxf(m_run[v], tm);
      alpha[v] = __expf(m_run[v] - mnew);
      m_run[v] = mnew;
    }

    // ---- P = exp(lm - m), row sums, spill P (bf16) for the PV WMMA ----
    float rs[8];
#pragma unroll
    for (int v = 0; v < 8; ++v) {
      float p = __expf(lm[v] - m_run[v]);
      Ps[(v + hi * 8) * TJ + keyBase + ln] = (bf16_t)p;
      rs[v] = p;
    }
#pragma unroll
    for (int m = 1; m < 16; m <<= 1)
#pragma unroll
      for (int v = 0; v < 8; ++v) rs[v] += __shfl_xor(rs[v], m, 32);
    if (ln == 0) {
#pragma unroll
      for (int v = 0; v < 8; ++v) RedS[wave * MT + v + hi * 8] = rs[v];
    }
    __syncthreads();

    // ---- l update + accumulator rescale ----
#pragma unroll
    for (int v = 0; v < 8; ++v) {
      int row = v + hi * 8;
      float ts = RedS[row];
#pragma unroll
      for (int w = 1; w < NWAVE; ++w) ts += RedS[w * MT + row];
      l_run[v] = l_run[v] * alpha[v] + ts;
    }
#pragma unroll
    for (int t = 0; t < 8; ++t)
#pragma unroll
      for (int v = 0; v < 8; ++v) o[t][v] *= alpha[v];

    // ---- PV: O(16 x 128 slice) += P(16x64) * V(64 x 128 slice) ----
#pragma unroll
    for (int nt = 0; nt < 8; ++nt) {
      int ncol = wave * 128 + nt * 16 + ln;               // B: N = ln
#pragma unroll
      for (int ks2 = 0; ks2 < 2; ++ks2) {
        v8bf a0 = *(const v8bf*)(Ps + ln * TJ + ks2 * 32 + hi * 16);
        v8bf a1 = *(const v8bf*)(Ps + ln * TJ + ks2 * 32 + hi * 16 + 8);
        v16bf A = __builtin_shufflevector(a0, a1,
            0,1,2,3,4,5,6,7,8,9,10,11,12,13,14,15);
        v16bf Bv = *(const v16bf*)(VtCur + (size_t)ncol * TJ +
                                   ks2 * 32 + hi * 16);
        o[nt] = __builtin_amdgcn_wmma_f32_16x16x32_bf16(false, A, false, Bv,
                                                        (short)0, o[nt],
                                                        false, false);
      }
    }
  }

  // ---- normalize and store fp32 output ----
#pragma unroll
  for (int t = 0; t < 8; ++t) {
    int colBase = wave * 128 + t * 16 + ln;
#pragma unroll
    for (int v = 0; v < 8; ++v) {
      float val = o[t][v] / l_run[v];
      outb[(size_t)(i0 + v + hi * 8) * D_DIM + colBase] = val;
    }
  }
}

extern "C" void kernel_launch(void* const* d_in, const int* in_sizes, int n_in,
                              void* d_out, int out_size, void* d_ws, size_t ws_size,
                              hipStream_t stream) {
  (void)in_sizes; (void)n_in; (void)out_size; (void)ws_size;
  const float* in = (const float*)d_in[0];
  float* out = (float*)d_out;
  bf16_t* in16 = (bf16_t*)d_ws;   // 4*4096*512*2 = 16.8MB of workspace

  // one-time precision downconvert (HBM/L2-bound, trivial at 23.3 TB/s)
  const int total = BATCH * S_LEN * D_DIM;
  hipLaunchKernelGGL(cvt_to_bf16, dim3(total / (256 * 8)), dim3(256), 0, stream,
                     in, in16);

  (void)hipFuncSetAttribute((const void*)sla_flash_wmma,
                            hipFuncAttributeMaxDynamicSharedMemorySize,
                            SMEM_BYTES);
  dim3 grid(BATCH * (S_LEN / MT));   // 1024 workgroups
  dim3 block(BLOCK);                 // 128 threads = 4 wave32
  hipLaunchKernelGGL(sla_flash_wmma, grid, block, SMEM_BYTES, stream, in16, out);
}
